// ModAttn_40905268527627
// MI455X (gfx1250) — matrix operand order repaired and checked by reading, not compiled, not run on hardware
//
#include <hip/hip_runtime.h>
#include <math.h>

// ---------------- Problem constants ----------------
#define DIN 512
#define N_TOK 1024
#define NFUNC 4
#define DCOND 128
#define H_HEADS 8
#define HD 64
#define NOUT_QKV 1536

typedef __bf16 bf16;
typedef __attribute__((ext_vector_type(16))) __bf16 v16bf;
typedef __attribute__((ext_vector_type(8)))  __bf16 v8bf;
typedef __attribute__((ext_vector_type(8)))  float  v8f;

// float -> bf16, round-to-nearest-even
__device__ __forceinline__ bf16 f2bf(float f) {
  union { float f; unsigned u; } v; v.f = f;
  unsigned r = v.u + 0x7FFFu + ((v.u >> 16) & 1u);
  unsigned short h = (unsigned short)(r >> 16);
  return __builtin_bit_cast(bf16, h);
}

// A fragment (16x32 bf16): lanes 0-15 row=lane, elems 0..15 = K k..k+15 contiguous;
// lanes 16-31 same rows with K offset +16 (caller folds the +16 into k).
__device__ __forceinline__ v16bf load_fragA(const bf16* rowptr, int k) {
  return *(const v16bf*)(rowptr + k);
}

// B fragment (32x16 bf16): lane holds column n = lane&15; lanes 0-15 hold K 0..15
// contiguous, lanes 16-31 hold K 16..31. colptr = row of B^T for column n; caller
// folds the +16 into k for lanes >= 16.
__device__ __forceinline__ v16bf load_fragB(const bf16* colptr, int k) {
  return *(const v16bf*)(colptr + k);
}

__device__ __forceinline__ v8f wmma_bf(v16bf a, v16bf b, v8f c) {
  return __builtin_amdgcn_wmma_f32_16x16x32_bf16(false, a, false, b, (short)0, c,
                                                 false, false);
}

// ---------------- code_mod: layernorm(w_c @ code) ----------------
__global__ void codemod_kernel(const float* __restrict__ w_c,
                               const float* __restrict__ code,
                               const float* __restrict__ gq, const float* __restrict__ bq,
                               const float* __restrict__ gp, const float* __restrict__ bp,
                               float* __restrict__ cmq, float* __restrict__ cmp) {
  int nf = blockIdx.x;        // 0..NFUNC-1
  int which = blockIdx.y;     // 0 = qkv, 1 = proj
  int d = threadIdx.x;        // 0..DIN-1
  float t = 0.f;
  for (int c = 0; c < DCOND; ++c) t += w_c[d * DCOND + c] * code[c * NFUNC + nf];
  __shared__ float red[DIN];
  __shared__ float s_mu, s_rs;
  red[d] = t; __syncthreads();
  for (int s = DIN / 2; s > 0; s >>= 1) { if (d < s) red[d] += red[d + s]; __syncthreads(); }
  if (d == 0) s_mu = red[0] * (1.0f / DIN);
  __syncthreads();
  float dv = t - s_mu;
  red[d] = dv * dv; __syncthreads();
  for (int s = DIN / 2; s > 0; s >>= 1) { if (d < s) red[d] += red[d + s]; __syncthreads(); }
  if (d == 0) s_rs = rsqrtf(red[0] * (1.0f / DIN) + 1e-5f);
  __syncthreads();
  const float* g = which ? gp : gq;
  const float* b = which ? bp : bq;
  float* o = which ? cmp : cmq;
  o[nf * DIN + d] = dv * s_rs * g[d] + b[d];
}

// ---------------- f32 -> bf16 convert ----------------
__global__ void convert_kernel(const float* __restrict__ src, bf16* __restrict__ dst, int n) {
  int i = blockIdx.x * 256 + threadIdx.x;
  if (i < n) dst[i] = f2bf(src[i]);
}

// ---------------- modulate (x * code_mod) and convert to bf16 ----------------
// bx=1: src is [N,DIN] broadcast over f; bx=0: src is [NFUNC,N,DIN]
__global__ void modulate_kernel(const float* __restrict__ src, const float* __restrict__ cm,
                                bf16* __restrict__ dst, int bx) {
  size_t i = (size_t)blockIdx.x * 256 + threadIdx.x;  // < NFUNC*N*DIN
  int d = (int)(i & (DIN - 1));
  size_t f = i / ((size_t)N_TOK * DIN);
  size_t si = bx ? (i % ((size_t)N_TOK * DIN)) : i;
  dst[i] = f2bf(src[si] * cm[f * DIN + d]);
}

// ---------------- compat[n][m] = sum_f comp[f][n]*comp[f][m] ----------------
__global__ void compat_kernel(const float* __restrict__ comp, float* __restrict__ out) {
  size_t i = (size_t)blockIdx.x * 256 + threadIdx.x;
  int n = (int)(i >> 10), m = (int)(i & 1023);
  float s = 0.f;
#pragma unroll
  for (int f = 0; f < NFUNC; ++f) s += comp[f * N_TOK + n] * comp[f * N_TOK + m];
  out[i] = s;
}

// store one 16x16 fp32 D-tile into qkv (q/k cols, row-major bf16) or vT (v cols,
// transposed bf16, contiguous 16B store)
__device__ __forceinline__ void store_qkv_tile(v8f c, int f, int mbase, int cg, int lane,
                                               const float* __restrict__ bias,
                                               bf16* __restrict__ qkv,
                                               bf16* __restrict__ vT) {
  float bv = bias[cg];
  int m0 = mbase + ((lane & 16) ? 8 : 0);
  if (cg < 2 * DIN) {
#pragma unroll
    for (int i = 0; i < 8; ++i)
      qkv[((size_t)f * N_TOK + m0 + i) * NOUT_QKV + cg] = f2bf(c[i] + bv);
  } else {
    v8bf o;
#pragma unroll
    for (int i = 0; i < 8; ++i) o[i] = f2bf(c[i] + bv);
    *(v8bf*)(vT + ((size_t)f * DIN + (cg - 2 * DIN)) * N_TOK + m0) = o;
  }
}

// ---------------- QKV GEMM: 32x32 macro-tile per wave (2x2 WMMA tiles) ----------------
__global__ __launch_bounds__(256) void gemm_qkv_kernel(
    const bf16* __restrict__ A, const bf16* __restrict__ W,
    const float* __restrict__ bias, bf16* __restrict__ qkv, bf16* __restrict__ vT) {
  int lane = threadIdx.x & 31, wv = threadIdx.x >> 5;
  int ct = blockIdx.x;                 // 0..47 column pairs
  int rt = blockIdx.y * 8 + wv;        // 0..31 row pairs
  int f  = blockIdx.z;
  int row = lane & 15;
  int kadjA = (lane & 16) ? 16 : 0;
  int kadjB = (lane & 16) ? 16 : 0;
  const bf16* arow0 = A + ((size_t)f * N_TOK + rt * 32 + row) * DIN;
  const bf16* arow1 = arow0 + (size_t)16 * DIN;
  const bf16* brow0 = W + (size_t)(ct * 32 + row) * DIN;
  const bf16* brow1 = brow0 + (size_t)16 * DIN;
  v8f c00 = {}, c01 = {}, c10 = {}, c11 = {};
  for (int k0 = 0; k0 < DIN; k0 += 32) {
    __builtin_prefetch(arow0 + k0 + 128, 0, 1);
    __builtin_prefetch(arow1 + k0 + 128, 0, 1);
    __builtin_prefetch(brow0 + k0 + 128, 0, 1);
    __builtin_prefetch(brow1 + k0 + 128, 0, 1);
    v16bf a0 = load_fragA(arow0, k0 + kadjA);
    v16bf a1 = load_fragA(arow1, k0 + kadjA);
    v16bf b0 = load_fragB(brow0, k0 + kadjB);
    v16bf b1 = load_fragB(brow1, k0 + kadjB);
    c00 = wmma_bf(a0, b0, c00);
    c01 = wmma_bf(a0, b1, c01);
    c10 = wmma_bf(a1, b0, c10);
    c11 = wmma_bf(a1, b1, c11);
  }
  int cg0 = ct * 32 + row, cg1 = cg0 + 16;
  store_qkv_tile(c00, f, rt * 32,      cg0, lane, bias, qkv, vT);
  store_qkv_tile(c01, f, rt * 32,      cg1, lane, bias, qkv, vT);
  store_qkv_tile(c10, f, rt * 32 + 16, cg0, lane, bias, qkv, vT);
  store_qkv_tile(c11, f, rt * 32 + 16, cg1, lane, bias, qkv, vT);
}

// ---------------- Attention: dual-softmax with compat modulation ----------------
__global__ __launch_bounds__(512) void attn_kernel(
    const bf16* __restrict__ qkv, const bf16* __restrict__ vT,
    const float* __restrict__ compat, float* __restrict__ yhat) {
  __shared__ float sc[16][N_TOK];      // 64 KB fp32 score strip (16 query rows)
  __shared__ bf16  wm[16][N_TOK];      // 32 KB bf16 normalized Wm (PV B-operand)
  __shared__ float pbuf[16][32][8];    // 16 KB PV partials
  int lane = threadIdx.x & 31, wv = threadIdx.x >> 5;
  int qb = blockIdx.x, f = blockIdx.y, h = blockIdx.z;
  int row = lane & 15;
  int kadjA = (lane & 16) ? 16 : 0;
  int kadjB = (lane & 16) ? 16 : 0;
  const float scale = 0.125f;          // HD^-0.5

  // Q fragments (16 x 64), loaded once per wave
  const bf16* qrow = qkv + ((size_t)f * N_TOK + qb * 16 + row) * NOUT_QKV + h * HD;
  v16bf qa0 = load_fragA(qrow, 0 + kadjA);
  v16bf qa1 = load_fragA(qrow, 32 + kadjA);

  // Phase 1: scores = Q K^T * scale ; 64 key tiles over 16 waves
  for (int t = 0; t < 4; ++t) {
    int mt = wv + t * 16;
    const bf16* krow = qkv + ((size_t)f * N_TOK + mt * 16 + row) * NOUT_QKV + DIN + h * HD;
    v16bf kb0 = load_fragB(krow, 0 + kadjB);
    v16bf kb1 = load_fragB(krow, 32 + kadjB);
    v8f c = {};
    c = wmma_bf(qa0, kb0, c);
    c = wmma_bf(qa1, kb1, c);
    int q0 = (lane & 16) ? 8 : 0;
#pragma unroll
    for (int i = 0; i < 8; ++i) sc[q0 + i][mt * 16 + row] = c[i] * scale;
  }
  __syncthreads();

  // Phase 2: wave wv owns query row wv: softmax -> *compat -> softmax -> bf16 Wm
  {
    float* srow = &sc[wv][0];
    int qg = qb * 16 + wv;
    float mx = -3.0e38f;
    for (int j = lane; j < N_TOK; j += 32) mx = fmaxf(mx, srow[j]);
#pragma unroll
    for (int s = 16; s; s >>= 1) mx = fmaxf(mx, __shfl_xor(mx, s, 32));
    float sm = 0.f;
    for (int j = lane; j < N_TOK; j += 32) { float e = __expf(srow[j] - mx); srow[j] = e; sm += e; }
#pragma unroll
    for (int s = 16; s; s >>= 1) sm += __shfl_xor(sm, s, 32);
    float inv = 1.0f / sm;
    const float* crow = compat + (size_t)qg * N_TOK;
    float mx2 = -3.0e38f;
    for (int j = lane; j < N_TOK; j += 32) {
      float w = srow[j] * inv * crow[j];
      srow[j] = w;
      mx2 = fmaxf(mx2, w);
    }
#pragma unroll
    for (int s = 16; s; s >>= 1) mx2 = fmaxf(mx2, __shfl_xor(mx2, s, 32));
    float sm2 = 0.f;
    for (int j = lane; j < N_TOK; j += 32) { float e = __expf(srow[j] - mx2); srow[j] = e; sm2 += e; }
#pragma unroll
    for (int s = 16; s; s >>= 1) sm2 += __shfl_xor(sm2, s, 32);
    float inv2 = 1.0f / sm2;
    for (int j = lane; j < N_TOK; j += 32) wm[wv][j] = f2bf(srow[j] * inv2);
  }
  __syncthreads();

  // Phase 3: y^T = V^T @ Wm^T ; waves = 4 d-tiles x 4 K-chunks
  int dt = wv & 3, kc = wv >> 2;
  const bf16* vrow = vT + ((size_t)f * DIN + h * HD + dt * 16 + row) * N_TOK;
  v8f c = {};
  for (int k0 = kc * 256; k0 < kc * 256 + 256; k0 += 32) {
    v16bf a = load_fragA(vrow, k0 + kadjA);
    v16bf b = *(const v16bf*)(&wm[row][k0 + kadjB]);  // B[m][q] = Wm[q][m]
    c = wmma_bf(a, b, c);
  }
#pragma unroll
  for (int i = 0; i < 8; ++i) pbuf[wv][lane][i] = c[i];
  __syncthreads();
  if (wv < 4) {                        // reduce 4 K-chunk partials for d-tile wv
    float acc[8];
#pragma unroll
    for (int i = 0; i < 8; ++i)
      acc[i] = pbuf[wv][lane][i] + pbuf[wv + 4][lane][i] +
               pbuf[wv + 8][lane][i] + pbuf[wv + 12][lane][i];
    int qg = qb * 16 + row;
    int d0 = wv * 16 + ((lane & 16) ? 8 : 0);
    float* orow = yhat + ((size_t)f * N_TOK + qg) * DIN + h * HD + d0;
#pragma unroll
    for (int i = 0; i < 8; ++i) orow[i] = acc[i];
  }
}

// ---------------- Proj GEMM: 32x32 macro-tile per wave -> out f32 ----------------
__global__ __launch_bounds__(256) void gemm_proj_kernel(
    const bf16* __restrict__ A, const bf16* __restrict__ W,
    const float* __restrict__ bias, float* __restrict__ out) {
  int lane = threadIdx.x & 31, wv = threadIdx.x >> 5;
  int ct = blockIdx.x;                 // 0..15 column pairs
  int rt = blockIdx.y * 8 + wv;        // 0..31 row pairs
  int f  = blockIdx.z;
  int row = lane & 15;
  int kadjA = (lane & 16) ? 16 : 0;
  int kadjB = (lane & 16) ? 16 : 0;
  const bf16* arow0 = A + ((size_t)f * N_TOK + rt * 32 + row) * DIN;
  const bf16* arow1 = arow0 + (size_t)16 * DIN;
  const bf16* brow0 = W + (size_t)(ct * 32 + row) * DIN;
  const bf16* brow1 = brow0 + (size_t)16 * DIN;
  v8f c00 = {}, c01 = {}, c10 = {}, c11 = {};
  for (int k0 = 0; k0 < DIN; k0 += 32) {
    __builtin_prefetch(arow0 + k0 + 128, 0, 1);
    __builtin_prefetch(arow1 + k0 + 128, 0, 1);
    __builtin_prefetch(brow0 + k0 + 128, 0, 1);
    __builtin_prefetch(brow1 + k0 + 128, 0, 1);
    v16bf a0 = load_fragA(arow0, k0 + kadjA);
    v16bf a1 = load_fragA(arow1, k0 + kadjA);
    v16bf b0 = load_fragB(brow0, k0 + kadjB);
    v16bf b1 = load_fragB(brow1, k0 + kadjB);
    c00 = wmma_bf(a0, b0, c00);
    c01 = wmma_bf(a0, b1, c01);
    c10 = wmma_bf(a1, b0, c10);
    c11 = wmma_bf(a1, b1, c11);
  }
  int cg0 = ct * 32 + row, cg1 = cg0 + 16;
  float bv0 = bias[cg0], bv1 = bias[cg1];
  int m0 = rt * 32 + ((lane & 16) ? 8 : 0);
#pragma unroll
  for (int i = 0; i < 8; ++i) {
    out[((size_t)f * N_TOK + m0 + i) * DIN + cg0] = c00[i] + bv0;
    out[((size_t)f * N_TOK + m0 + i) * DIN + cg1] = c01[i] + bv1;
    out[((size_t)f * N_TOK + m0 + 16 + i) * DIN + cg0] = c10[i] + bv0;
    out[((size_t)f * N_TOK + m0 + 16 + i) * DIN + cg1] = c11[i] + bv1;
  }
}

// ---------------- host launcher ----------------
extern "C" void kernel_launch(void* const* d_in, const int* in_sizes, int n_in,
                              void* d_out, int out_size, void* d_ws, size_t ws_size,
                              hipStream_t stream) {
  (void)in_sizes; (void)n_in; (void)out_size; (void)ws_size;
  const float* x        = (const float*)d_in[0];
  const float* compt    = (const float*)d_in[1];
  const float* code     = (const float*)d_in[2];
  const float* w_c      = (const float*)d_in[3];
  const float* W_qkv    = (const float*)d_in[4];
  const float* b_qkv    = (const float*)d_in[5];
  const float* W_proj   = (const float*)d_in[6];
  const float* b_proj   = (const float*)d_in[7];
  const float* ln_qkv_g = (const float*)d_in[8];
  const float* ln_qkv_b = (const float*)d_in[9];
  const float* ln_pr_g  = (const float*)d_in[10];
  const float* ln_pr_b  = (const float*)d_in[11];

  char* wsp = (char*)d_ws;
  size_t off = 0;
  auto carve = [&](size_t bytes) -> char* {
    char* p = wsp + off;
    off += (bytes + 255) & ~(size_t)255;
    return p;
  };
  float* cmq    = (float*)carve((size_t)NFUNC * DIN * 4);
  float* cmp    = (float*)carve((size_t)NFUNC * DIN * 4);
  bf16*  wqkvb  = (bf16*) carve((size_t)NOUT_QKV * DIN * 2);
  bf16*  wprojb = (bf16*) carve((size_t)DIN * DIN * 2);
  bf16*  xmod   = (bf16*) carve((size_t)NFUNC * N_TOK * DIN * 2);
  bf16*  qkvb   = (bf16*) carve((size_t)NFUNC * N_TOK * NOUT_QKV * 2);
  bf16*  vT     = (bf16*) carve((size_t)NFUNC * DIN * N_TOK * 2);
  float* compatW= (float*)carve((size_t)N_TOK * N_TOK * 4);
  float* yhat   = (float*)carve((size_t)NFUNC * N_TOK * DIN * 4);
  bf16*  ymod   = (bf16*) carve((size_t)NFUNC * N_TOK * DIN * 2);

  codemod_kernel<<<dim3(NFUNC, 2), DIN, 0, stream>>>(
      w_c, code, ln_qkv_g, ln_qkv_b, ln_pr_g, ln_pr_b, cmq, cmp);

  convert_kernel<<<(NOUT_QKV * DIN) / 256, 256, 0, stream>>>(W_qkv, wqkvb, NOUT_QKV * DIN);
  convert_kernel<<<(DIN * DIN) / 256, 256, 0, stream>>>(W_proj, wprojb, DIN * DIN);

  modulate_kernel<<<(NFUNC * N_TOK * DIN) / 256, 256, 0, stream>>>(x, cmq, xmod, 1);

  compat_kernel<<<(N_TOK * N_TOK) / 256, 256, 0, stream>>>(compt, compatW);

  // 2x2 tiles: 48 column pairs x (32 row pairs / 8 waves) x NFUNC
  gemm_qkv_kernel<<<dim3(NOUT_QKV / 32, N_TOK / 32 / 8, NFUNC), 256, 0, stream>>>(
      xmod, wqkvb, b_qkv, qkvb, vT);

  attn_kernel<<<dim3(N_TOK / 16, NFUNC, H_HEADS), 512, 0, stream>>>(
      qkvb, vT, compatW, yhat);

  modulate_kernel<<<(NFUNC * N_TOK * DIN) / 256, 256, 0, stream>>>(yhat, cmp, ymod, 0);

  gemm_proj_kernel<<<dim3(DIN / 32, N_TOK / 32 / 8, NFUNC), 256, 0, stream>>>(
      ymod, wprojb, b_proj, (float*)d_out);
}
